// Attention_40767829574450
// MI455X (gfx1250) — compile-verified, matrix operand order checked by reading, not run on hardware
//
#include <hip/hip_runtime.h>

#define BB 16
#define TXv 512
#define TYv 1600
#define MELv 80
#define MELP 96
#define Hv 192
#define FFv 768
#define NHv 2
#define DHv 96
#define NLv 6
#define SPKv 64
#define GSTv 256
#define NEGF (-1e9f)

typedef __attribute__((ext_vector_type(16))) _Float16 v16h;
typedef __attribute__((ext_vector_type(8)))  float    v8f;

struct I4 { int x, y, z, w; };
union Frag { v16h v; I4 q[2]; };

// ---------------------------------------------------------------------------
// Generic batched WMMA GEMM:  C[z] = alpha * A[z] (MxK, f16 row-major)
//                                   @ B[z]^T (B stored N-major: (N,K) f16)
//                             + colAdd[n] + rowAdd[m]
// batch z decomposed as z = zo*innerCount + zi with separate strides.
// 4 waves/block arranged 2x2; each wave owns a 32x32 register tile
// (2x2 16x16 accumulators) -> 4 WMMAs per K-step per wave.
// Edge 16-tiles are clamped (recompute same values) so M,N only need to be
// multiples of 16.
// ---------------------------------------------------------------------------
__global__ __launch_bounds__(128)
void k_gemm(const _Float16* __restrict__ A, int lda, long long aSO, long long aSI,
            const _Float16* __restrict__ Bm, int ldb, long long bSO, long long bSI,
            float* __restrict__ C, int ldc, long long cSO, long long cSI,
            int innerCount, int M, int N, int K, float alpha,
            const float* __restrict__ colAdd, long long colStride,
            const float* __restrict__ rowAdd, long long rowStride)
{
    int z  = blockIdx.z;
    int zo = z / innerCount, zi = z - zo * innerCount;
    A  += zo * aSO + zi * aSI;
    Bm += zo * bSO + zi * bSI;
    C  += zo * cSO + zi * cSI;

    int wave = threadIdx.x >> 5, lane = threadIdx.x & 31;
    int m0 = blockIdx.y * 64 + (wave >> 1) * 32;
    int n0 = blockIdx.x * 64 + (wave & 1) * 32;
    if (m0 >= M || n0 >= N) return;

    int r16 = lane & 15, hi = lane >> 4, koff = hi * 8;
    int mA0 = m0, mA1 = (m0 + 16 < M) ? (m0 + 16) : (M - 16);
    int nB0 = n0, nB1 = (n0 + 16 < N) ? (n0 + 16) : (N - 16);

    const _Float16* ap0 = A  + (size_t)(mA0 + r16) * lda + koff;
    const _Float16* ap1 = A  + (size_t)(mA1 + r16) * lda + koff;
    const _Float16* bp0 = Bm + (size_t)(nB0 + r16) * ldb + koff;
    const _Float16* bp1 = Bm + (size_t)(nB1 + r16) * ldb + koff;

    v8f acc00 = {0.f,0.f,0.f,0.f,0.f,0.f,0.f,0.f};
    v8f acc01 = {0.f,0.f,0.f,0.f,0.f,0.f,0.f,0.f};
    v8f acc10 = {0.f,0.f,0.f,0.f,0.f,0.f,0.f,0.f};
    v8f acc11 = {0.f,0.f,0.f,0.f,0.f,0.f,0.f,0.f};

    for (int k0 = 0; k0 < K; k0 += 32) {
        Frag a0, a1, b0, b1;
        a0.q[0] = *(const I4*)(ap0 + k0);
        a0.q[1] = *(const I4*)(ap0 + k0 + 16);
        a1.q[0] = *(const I4*)(ap1 + k0);
        a1.q[1] = *(const I4*)(ap1 + k0 + 16);
        b0.q[0] = *(const I4*)(bp0 + k0);
        b0.q[1] = *(const I4*)(bp0 + k0 + 16);
        b1.q[0] = *(const I4*)(bp1 + k0);
        b1.q[1] = *(const I4*)(bp1 + k0 + 16);
        acc00 = __builtin_amdgcn_wmma_f32_16x16x32_f16(
            false, a0.v, false, b0.v, (short)0, acc00, false, false);
        acc01 = __builtin_amdgcn_wmma_f32_16x16x32_f16(
            false, a0.v, false, b1.v, (short)0, acc01, false, false);
        acc10 = __builtin_amdgcn_wmma_f32_16x16x32_f16(
            false, a1.v, false, b0.v, (short)0, acc10, false, false);
        acc11 = __builtin_amdgcn_wmma_f32_16x16x32_f16(
            false, a1.v, false, b1.v, (short)0, acc11, false, false);
    }

    float cadd0 = colAdd ? colAdd[z * colStride + nB0 + r16] : 0.0f;
    float cadd1 = colAdd ? colAdd[z * colStride + nB1 + r16] : 0.0f;
#pragma unroll
    for (int r = 0; r < 8; ++r) {
        int mr0 = mA0 + hi * 8 + r;
        int mr1 = mA1 + hi * 8 + r;
        float ra0 = rowAdd ? rowAdd[z * rowStride + mr0] : 0.0f;
        float ra1 = rowAdd ? rowAdd[z * rowStride + mr1] : 0.0f;
        C[(size_t)mr0 * ldc + nB0 + r16] = alpha * acc00[r] + cadd0 + ra0;
        C[(size_t)mr0 * ldc + nB1 + r16] = alpha * acc01[r] + cadd1 + ra0;
        C[(size_t)mr1 * ldc + nB0 + r16] = alpha * acc10[r] + cadd0 + ra1;
        C[(size_t)mr1 * ldc + nB1 + r16] = alpha * acc11[r] + cadd1 + ra1;
    }
}

// ---------------------------------------------------------------------------
// Small elementwise / prep kernels
// ---------------------------------------------------------------------------
__global__ void k_cond(const float* __restrict__ spkr, const float* __restrict__ gst,
                       const float* __restrict__ w_spk, const float* __restrict__ w_gst,
                       float* __restrict__ cond)
{
    int i = blockIdx.x * blockDim.x + threadIdx.x;     // B*H
    if (i >= BB * Hv) return;
    int c = i % Hv, b = i / Hv;
    float s = 0.f;
    for (int k = 0; k < SPKv; k++) s += spkr[b * SPKv + k] * w_spk[k * Hv + c];
    for (int k = 0; k < GSTv; k++) s += gst[b * GSTv + k] * w_gst[k * Hv + c];
    cond[i] = s;
}

__global__ void k_embed(const int* __restrict__ text, const float* __restrict__ emb,
                        const float* __restrict__ cond, const int* __restrict__ tlen,
                        float* __restrict__ X)
{
    long long i = (long long)blockIdx.x * blockDim.x + threadIdx.x;  // B*TX*H
    if (i >= (long long)BB * TXv * Hv) return;
    int c = (int)(i % Hv);
    long long bt = i / Hv;
    int t = (int)(bt % TXv), b = (int)(bt / TXv);
    float m = (t < tlen[b]) ? 1.0f : 0.0f;
    int tok = text[bt];
    X[i] = (emb[(size_t)tok * Hv + c] * 13.856406460551018f + cond[b * Hv + c]) * m;
}

__global__ void k_cvt_f16(const float* __restrict__ src, _Float16* __restrict__ dst,
                          long long n)
{
    long long i = (long long)blockIdx.x * blockDim.x + threadIdx.x;
    if (i < n) dst[i] = (_Float16)src[i];
}

__global__ void k_transpose_f16(const float* __restrict__ src, _Float16* __restrict__ dst,
                                int K, int N)   // src (K,N) -> dst (N,K)
{
    long long i = (long long)blockIdx.x * blockDim.x + threadIdx.x;
    if (i >= (long long)K * N) return;
    int k = (int)(i % K), n = (int)(i / K);
    dst[(size_t)n * K + k] = (_Float16)src[(size_t)k * N + n];
}

__global__ void k_vt(const float* __restrict__ V, _Float16* __restrict__ VT)
{   // V (B*TX, H) -> VT[((b*NH+h)*DH + d)*TX + t]
    long long i = (long long)blockIdx.x * blockDim.x + threadIdx.x;
    if (i >= (long long)BB * NHv * DHv * TXv) return;
    int t = (int)(i % TXv);
    long long r = i / TXv;
    int d = (int)(r % DHv); r /= DHv;
    int h = (int)(r % NHv); int b = (int)(r / NHv);
    VT[i] = (_Float16)V[((size_t)b * TXv + t) * Hv + h * DHv + d];
}

__global__ __launch_bounds__(256)
void k_softmax(const float* __restrict__ S, _Float16* __restrict__ P,
               const int* __restrict__ tlen)
{
    long long row = blockIdx.x;                 // [0, B*NH*TX)
    int q = (int)(row % TXv);
    int b = (int)(row / ((long long)TXv * NHv));
    int len = tlen[b];
    bool qv = q < len;
    const float* s = S + row * TXv;
    int tid = threadIdx.x, lane = tid & 31, wave = tid >> 5;
    __shared__ float sm[8];
    __shared__ float bc[2];
    int c0 = tid, c1 = tid + 256;
    float v0 = (qv && c0 < len) ? s[c0] : NEGF;
    float v1 = (qv && c1 < len) ? s[c1] : NEGF;
    float mx = fmaxf(v0, v1);
    for (int o = 16; o > 0; o >>= 1) mx = fmaxf(mx, __shfl_xor(mx, o, 32));
    if (lane == 0) sm[wave] = mx;
    __syncthreads();
    if (tid == 0) { float m = sm[0]; for (int i = 1; i < 8; i++) m = fmaxf(m, sm[i]); bc[0] = m; }
    __syncthreads();
    mx = bc[0];
    float e0 = __expf(v0 - mx), e1 = __expf(v1 - mx);
    float ss = e0 + e1;
    for (int o = 16; o > 0; o >>= 1) ss += __shfl_xor(ss, o, 32);
    if (lane == 0) sm[wave] = ss;
    __syncthreads();
    if (tid == 0) { float t = 0.f; for (int i = 0; i < 8; i++) t += sm[i]; bc[1] = t; }
    __syncthreads();
    float inv = 1.0f / bc[1];
    P[row * TXv + c0] = (_Float16)(e0 * inv);
    P[row * TXv + c1] = (_Float16)(e1 * inv);
}

__global__ __launch_bounds__(128)
void k_resid_ln(float* __restrict__ X, const float* __restrict__ D,
                const float* __restrict__ g, const float* __restrict__ be,
                const int* __restrict__ tlen, int applyMask)
{
    int wave = threadIdx.x >> 5, lane = threadIdx.x & 31;
    long long row = (long long)blockIdx.x * 4 + wave;    // [0, B*TX)
    float* xr = X + row * Hv;
    const float* dr = D + row * Hv;
    float v[6]; float s = 0.f, s2 = 0.f;
#pragma unroll
    for (int i = 0; i < 6; i++) {
        int c = lane + i * 32;
        v[i] = xr[c] + dr[c];
        s += v[i]; s2 += v[i] * v[i];
    }
    for (int o = 16; o > 0; o >>= 1) { s += __shfl_xor(s, o, 32); s2 += __shfl_xor(s2, o, 32); }
    float m  = s / (float)Hv;
    float var = s2 / (float)Hv - m * m;
    float rs = rsqrtf(var + 1e-5f);
    float mask = 1.0f;
    if (applyMask) {
        int t = (int)(row % TXv), b = (int)(row / TXv);
        mask = (t < tlen[b]) ? 1.0f : 0.0f;
    }
#pragma unroll
    for (int i = 0; i < 6; i++) {
        int c = lane + i * 32;
        xr[c] = ((v[i] - m) * rs * g[c] + be[c]) * mask;
    }
}

__global__ void k_im2col3(const float* __restrict__ X, _Float16* __restrict__ D,
                          const int* __restrict__ tlen, int C, int doRelu)
{
    long long i = (long long)blockIdx.x * blockDim.x + threadIdx.x;
    long long total = (long long)BB * TXv * 3 * C;
    if (i >= total) return;
    int j = (int)(i % (3 * C));
    long long bt = i / (3 * C);
    int t = (int)(bt % TXv), b = (int)(bt / TXv);
    int k = j / C, c = j - k * C;
    int ts = t + k - 1;
    float v = 0.f;
    if (ts >= 0 && ts < TXv && ts < tlen[b]) {
        v = X[((size_t)b * TXv + ts) * C + c];
        if (doRelu) v = fmaxf(v, 0.f);
    }
    D[i] = (_Float16)v;
}

__global__ void k_keypad(const float* __restrict__ KEY, _Float16* __restrict__ KH,
                         const int* __restrict__ tlen)
{   // KEY (B*TX, 80) -> KH (B*TX, 96), masked by t < tlen[b]
    long long i = (long long)blockIdx.x * blockDim.x + threadIdx.x;
    if (i >= (long long)BB * TXv * MELP) return;
    int d = (int)(i % MELP);
    long long bt = i / MELP;
    int t = (int)(bt % TXv), b = (int)(bt / TXv);
    float v = 0.f;
    if (d < MELv && t < tlen[b]) v = KEY[bt * MELv + d];
    KH[i] = (_Float16)v;
}

__global__ void k_specpad(const float* __restrict__ SP, _Float16* __restrict__ SH)
{   // SP (B*TY, 80) -> SH (B*TY, 96)
    long long i = (long long)blockIdx.x * blockDim.x + threadIdx.x;
    if (i >= (long long)BB * TYv * MELP) return;
    int d = (int)(i % MELP);
    long long by = i / MELP;
    SH[i] = (_Float16)((d < MELv) ? SP[by * MELv + d] : 0.f);
}

__global__ void k_kn(const _Float16* __restrict__ KH, float* __restrict__ kn)
{
    int i = blockIdx.x * blockDim.x + threadIdx.x;      // B*TX
    if (i >= BB * TXv) return;
    float s = 0.f;
    for (int d = 0; d < MELP; d++) { float v = (float)KH[(size_t)i * MELP + d]; s += v * v; }
    kn[i] = -0.5f * s;
}

__global__ void k_sn(const float* __restrict__ SP, float* __restrict__ sn)
{
    int i = blockIdx.x * blockDim.x + threadIdx.x;      // B*TY
    if (i >= BB * TYv) return;
    float s = 0.f;
    for (int d = 0; d < MELv; d++) { float v = SP[(size_t)i * MELv + d]; s += v * v; }
    sn[i] = -0.5f * s;
}

// ---------------------------------------------------------------------------
// Monotonic alignment DP (serial in y, column relayed through LDS)
// ---------------------------------------------------------------------------
__global__ __launch_bounds__(512)
void k_dp(const float* __restrict__ LL, float* __restrict__ DP,
          const int* __restrict__ tlen, const int* __restrict__ slen)
{
    int b = blockIdx.x, x = threadIdx.x;
    int tx = tlen[b], tyr = slen[b] / 2;
    __shared__ float prev[TXv];
    const float* ll = LL + (size_t)b * TXv * TYv;
    float* dp = DP + (size_t)b * TXv * TYv;
    bool xv = x < tx;
    float cur = (x == 0) ? ((xv && 0 < tyr) ? ll[(size_t)x * TYv] : 0.f) : NEGF;
    dp[(size_t)x * TYv] = cur;
    prev[x] = cur;
    __syncthreads();
    for (int y = 1; y < TYv; y++) {
        float p  = prev[x];
        float sh = (x > 0) ? prev[x - 1] : NEGF;
        float v  = (xv && y < tyr) ? ll[(size_t)x * TYv + y] : 0.f;
        cur = v + fmaxf(p, sh);
        __syncthreads();
        prev[x] = cur;
        dp[(size_t)x * TYv + y] = cur;
        __syncthreads();
    }
}

__global__ void k_fill_out(float* __restrict__ out, const int* __restrict__ tlen,
                           const int* __restrict__ slen)
{
    const long long AT = (long long)BB * TXv * TYv;
    long long i = (long long)blockIdx.x * blockDim.x + threadIdx.x;
    if (i >= AT) return;
    out[i] = 0.0f;                                   // attention zero-init
    int y = (int)(i % TYv);
    long long t = i / TYv;
    int x = (int)(t % TXv), b = (int)(t / TXv);
    out[AT + 1 + i] = (x < tlen[b] && y < slen[b] / 2) ? 1.0f : 0.0f;   // att_mask
}

__global__ __launch_bounds__(32)
void k_backtrack(const float* __restrict__ LL, const float* __restrict__ DP,
                 const int* __restrict__ tlen, const int* __restrict__ slen,
                 float* __restrict__ out)
{
    const long long AT = (long long)BB * TXv * TYv;
    __shared__ float ns[BB];
    int b = threadIdx.x;
    if (b < BB) {
        int tx = tlen[b], tyr = slen[b] / 2;
        const float* ll = LL + (size_t)b * TXv * TYv;
        const float* dp = DP + (size_t)b * TXv * TYv;
        float* att = out + (size_t)b * TXv * TYv;
        int index = tx - 1;
        float acc = 0.f;
        for (int y = TYv - 1; y >= 0; y--) {
            bool active = y < tyr;
            if (active) {
                att[(size_t)index * TYv + y] = 1.0f;
                acc += -ll[(size_t)index * TYv + y];
            }
            int ym = (y - 1 < 0) ? 0 : (y - 1);
            int im = (index - 1 < 0) ? 0 : (index - 1);
            float vc = dp[(size_t)index * TYv + ym];
            float vp = dp[(size_t)im * TYv + ym];
            bool move = (index != 0) && ((index == y) || (vc < vp)) && active && (y > 0);
            index -= move ? 1 : 0;
        }
        ns[b] = acc / (float)tyr;
    }
    __syncthreads();
    if (threadIdx.x == 0) {
        float s = 0.f;
        for (int i = 0; i < BB; i++) s += ns[i];
        float nll = s / (float)BB;
        out[AT] = nll;
        out[2 * AT + 1] = nll;
    }
}

// ---------------------------------------------------------------------------
// Host orchestration
// ---------------------------------------------------------------------------
static inline long long cdiv(long long a, long long b) { return (a + b - 1) / b; }

extern "C" void kernel_launch(void* const* d_in, const int* in_sizes, int n_in,
                              void* d_out, int out_size, void* d_ws, size_t ws_size,
                              hipStream_t stream)
{
    (void)in_sizes; (void)n_in; (void)out_size; (void)ws_size;
    const int*   text   = (const int*)  d_in[0];
    const float* spec   = (const float*)d_in[1];
    const int*   tlen   = (const int*)  d_in[2];
    const int*   slen   = (const int*)  d_in[3];
    const float* spkr   = (const float*)d_in[4];
    const float* gst    = (const float*)d_in[5];
    const float* emb    = (const float*)d_in[6];
    const float* w_spk  = (const float*)d_in[7];
    const float* w_gst  = (const float*)d_in[8];
    const float* proj_w = (const float*)d_in[9];
    const float* proj_b = (const float*)d_in[10];
    float* out = (float*)d_out;

    // Workspace layout
    char* base = (char*)d_ws;
    size_t off = 0;
    auto take = [&](size_t bytes) { void* p = base + off; off += (bytes + 255) & ~(size_t)255; return p; };
    const long long ROWS = (long long)BB * TXv;            // 8192
    float*    cond   = (float*)   take(BB * Hv * 4);
    float*    X      = (float*)   take(ROWS * Hv * 4);
    _Float16* xh     = (_Float16*)take(ROWS * Hv * 2);
    _Float16* wT     = (_Float16*)take((size_t)3 * FFv * Hv * 2);   // max weightT
    float*    tmp    = (float*)   take(ROWS * FFv * 4);
    _Float16* q_h    = (_Float16*)take(ROWS * Hv * 2);
    _Float16* k_h    = (_Float16*)take(ROWS * Hv * 2);
    _Float16* vt_h   = (_Float16*)take(ROWS * Hv * 2);
    _Float16* P_h    = (_Float16*)take((size_t)BB * NHv * TXv * TXv * 2);
    char*     bigraw = (char*)    take((size_t)ROWS * 3 * FFv * 2); // S f32 / im2col alias
    float*    Sbuf   = (float*)bigraw;
    _Float16* imcol  = (_Float16*)bigraw;
    _Float16* spec_h = (_Float16*)take((size_t)BB * TYv * MELP * 2);
    _Float16* key_h  = (_Float16*)take(ROWS * MELP * 2);
    float*    kn     = (float*)   take(ROWS * 4);
    float*    sn     = (float*)   take((size_t)BB * TYv * 4);
    float*    LL     = (float*)   take((size_t)BB * TXv * TYv * 4);
    float*    DP     = (float*)   take((size_t)BB * TXv * TYv * 4);

    const float inv_sqrt_dh = 0.1020620726159657f;  // 1/sqrt(96)

    // ---- embedding + conditioning ----
    k_cond<<<dim3(cdiv(BB * Hv, 128)), dim3(128), 0, stream>>>(spkr, gst, w_spk, w_gst, cond);
    k_embed<<<dim3(cdiv(ROWS * Hv, 256)), dim3(256), 0, stream>>>(text, emb, cond, tlen, X);

    auto gemm = [&](const _Float16* A, int lda, long long aSO, long long aSI,
                    const _Float16* Bm, int ldb, long long bSO, long long bSI,
                    float* C, int ldc, long long cSO, long long cSI,
                    int inner, int M, int N, int K, float alpha,
                    const float* ca, long long cs, const float* ra, long long rs2, int batch) {
        dim3 g((unsigned)cdiv(N, 64), (unsigned)cdiv(M, 64), (unsigned)batch);
        k_gemm<<<g, dim3(128), 0, stream>>>(A, lda, aSO, aSI, Bm, ldb, bSO, bSI,
                                            C, ldc, cSO, cSI, inner, M, N, K, alpha,
                                            ca, cs, ra, rs2);
    };
    auto trans = [&](const float* src, int K, int N) {
        k_transpose_f16<<<dim3(cdiv((long long)K * N, 256)), dim3(256), 0, stream>>>(src, wT, K, N);
    };
    auto cvt = [&](const float* src, _Float16* dst, long long n) {
        k_cvt_f16<<<dim3(cdiv(n, 256)), dim3(256), 0, stream>>>(src, dst, n);
    };

    // ---- encoder layers ----
    for (int L = 0; L < NLv; L++) {
        int pb = 11 + 12 * L;
        const float* wq  = (const float*)d_in[pb + 0];
        const float* wk  = (const float*)d_in[pb + 1];
        const float* wv  = (const float*)d_in[pb + 2];
        const float* wo  = (const float*)d_in[pb + 3];
        const float* g1  = (const float*)d_in[pb + 4];
        const float* b1  = (const float*)d_in[pb + 5];
        const float* c1w = (const float*)d_in[pb + 6];
        const float* c1b = (const float*)d_in[pb + 7];
        const float* c2w = (const float*)d_in[pb + 8];
        const float* c2b = (const float*)d_in[pb + 9];
        const float* g2  = (const float*)d_in[pb + 10];
        const float* b2  = (const float*)d_in[pb + 11];

        cvt(X, xh, ROWS * Hv);
        // Q
        trans(wq, Hv, Hv);
        gemm(xh, Hv, 0, 0, wT, Hv, 0, 0, tmp, Hv, 0, 0, 1, (int)ROWS, Hv, Hv, 1.f,
             nullptr, 0, nullptr, 0, 1);
        cvt(tmp, q_h, ROWS * Hv);
        // K
        trans(wk, Hv, Hv);
        gemm(xh, Hv, 0, 0, wT, Hv, 0, 0, tmp, Hv, 0, 0, 1, (int)ROWS, Hv, Hv, 1.f,
             nullptr, 0, nullptr, 0, 1);
        cvt(tmp, k_h, ROWS * Hv);
        // V (transposed per head)
        trans(wv, Hv, Hv);
        gemm(xh, Hv, 0, 0, wT, Hv, 0, 0, tmp, Hv, 0, 0, 1, (int)ROWS, Hv, Hv, 1.f,
             nullptr, 0, nullptr, 0, 1);
        k_vt<<<dim3(cdiv(ROWS * Hv, 256)), dim3(256), 0, stream>>>(tmp, vt_h);
        // scores S = Q K^T / sqrt(dh), batch over (b,h)
        gemm(q_h, Hv, (long long)TXv * Hv, DHv,
             k_h, Hv, (long long)TXv * Hv, DHv,
             Sbuf, TXv, (long long)NHv * TXv * TXv, (long long)TXv * TXv,
             NHv, TXv, TXv, DHv, inv_sqrt_dh, nullptr, 0, nullptr, 0, BB * NHv);
        // softmax -> P (f16)
        k_softmax<<<dim3(BB * NHv * TXv), dim3(256), 0, stream>>>(Sbuf, P_h, tlen);
        // O = P @ V
        gemm(P_h, TXv, (long long)NHv * TXv * TXv, (long long)TXv * TXv,
             vt_h, TXv, (long long)NHv * DHv * TXv, (long long)DHv * TXv,
             tmp, Hv, (long long)TXv * Hv, DHv,
             NHv, TXv, DHv, TXv, 1.f, nullptr, 0, nullptr, 0, BB * NHv);
        cvt(tmp, xh, ROWS * Hv);
        // O @ Wo
        trans(wo, Hv, Hv);
        gemm(xh, Hv, 0, 0, wT, Hv, 0, 0, tmp, Hv, 0, 0, 1, (int)ROWS, Hv, Hv, 1.f,
             nullptr, 0, nullptr, 0, 1);
        // x = LN1(x + o)
        k_resid_ln<<<dim3((unsigned)(ROWS / 4)), dim3(128), 0, stream>>>(X, tmp, g1, b1, tlen, 0);
        // conv1: im2col(x * mask) @ W1 + b1c
        k_im2col3<<<dim3(cdiv(ROWS * 3 * Hv, 256)), dim3(256), 0, stream>>>(X, imcol, tlen, Hv, 0);
        trans(c1w, 3 * Hv, FFv);
        gemm(imcol, 3 * Hv, 0, 0, wT, 3 * Hv, 0, 0, tmp, FFv, 0, 0, 1,
             (int)ROWS, FFv, 3 * Hv, 1.f, c1b, 0, nullptr, 0, 1);
        // conv2: im2col(relu(y) * mask) @ W2 + b2c
        k_im2col3<<<dim3(cdiv(ROWS * 3 * FFv, 256)), dim3(256), 0, stream>>>(tmp, imcol, tlen, FFv, 1);
        trans(c2w, 3 * FFv, Hv);
        gemm(imcol, 3 * FFv, 0, 0, wT, 3 * FFv, 0, 0, tmp, Hv, 0, 0, 1,
             (int)ROWS, Hv, 3 * FFv, 1.f, c2b, 0, nullptr, 0, 1);
        // x = LN2(x + y) * mask
        k_resid_ln<<<dim3((unsigned)(ROWS / 4)), dim3(128), 0, stream>>>(X, tmp, g2, b2, tlen, 1);
    }

    // ---- projection to mel key ----
    cvt(X, xh, ROWS * Hv);
    trans(proj_w, Hv, MELv);
    gemm(xh, Hv, 0, 0, wT, Hv, 0, 0, tmp, MELv, 0, 0, 1, (int)ROWS, MELv, Hv, 1.f,
         proj_b, 0, nullptr, 0, 1);
    k_keypad<<<dim3(cdiv(ROWS * MELP, 256)), dim3(256), 0, stream>>>(tmp, key_h, tlen);
    k_specpad<<<dim3(cdiv((long long)BB * TYv * MELP, 256)), dim3(256), 0, stream>>>(spec, spec_h);
    k_kn<<<dim3(cdiv(ROWS, 256)), dim3(256), 0, stream>>>(key_h, kn);
    k_sn<<<dim3(cdiv((long long)BB * TYv, 256)), dim3(256), 0, stream>>>(spec, sn);

    // ---- loglik = key.spec - 0.5||key||^2 - 0.5||spec||^2 (batched WMMA GEMM) ----
    gemm(key_h, MELP, (long long)TXv * MELP, 0,
         spec_h, MELP, (long long)TYv * MELP, 0,
         LL, TYv, (long long)TXv * TYv, 0,
         1, TXv, TYv, MELP, 1.f, sn, TYv, kn, TXv, BB);

    // ---- DP, outputs ----
    k_dp<<<dim3(BB), dim3(TXv), 0, stream>>>(LL, DP, tlen, slen);
    k_fill_out<<<dim3(cdiv((long long)BB * TXv * TYv, 256)), dim3(256), 0, stream>>>(out, tlen, slen);
    k_backtrack<<<dim3(1), dim3(32), 0, stream>>>(LL, DP, tlen, slen, out);
}
